// NoPropDTEncoder_65360812311338
// MI455X (gfx1250) — matrix operand order repaired and checked by reading, not compiled, not run on hardware
//
#include <hip/hip_runtime.h>

// ---------------------------------------------------------------------------
// MI455X (gfx1250, wave32) implementation.
//
// Sizes: B=256, S=512, D=768, H=8, DH=96, T=4, MODES=16, NCLS=14.
//
// Strategy (see analysis): attention is collapsed to two streaming passes over
// x_feat per trunk step (HBM-bound, ~23.3TB/s roofline); all dense projection
// GEMMs run on v_wmma_f32_16x16x32_f16 with fp16 A/B and fp32 accumulate.
// FFT filter is a rank-16 projection/reconstruction (filt is scalar/mode).
// ---------------------------------------------------------------------------

#define D_     768
#define H_     8
#define DH_    96
#define T_     4
#define MODES_ 16
#define NCLS_  14
#define B_     256
#define S_     512

typedef __attribute__((ext_vector_type(16))) _Float16 v16h;
typedef __attribute__((ext_vector_type(8)))  _Float16 v8h;
typedef __attribute__((ext_vector_type(8)))  float    v8f;

union V16U { v16h v; v8h h[2]; };

__device__ __forceinline__ float warp_sum(float x) {
#pragma unroll
  for (int off = 16; off > 0; off >>= 1) x += __shfl_xor(x, off, 32);
  return x;
}
__device__ __forceinline__ float warp_max(float x) {
#pragma unroll
  for (int off = 16; off > 0; off >>= 1) x = fmaxf(x, __shfl_xor(x, off, 32));
  return x;
}
__device__ __forceinline__ float block_sum256(float x, float* sred) {
  x = warp_sum(x);
  const int wave = threadIdx.x >> 5, lane = threadIdx.x & 31;
  if (lane == 0) sred[wave] = x;
  __syncthreads();
  float t = 0.f;
#pragma unroll
  for (int i = 0; i < 8; ++i) t += sred[i];
  __syncthreads();
  return t;
}

// ---------------------------------------------------------------------------
// WMMA GEMM: C[M,N] = epilogue( A_f16[M,K] @ Bt_f16[N,K]^T ) with optional
// batch (grid.z) offsets. One wave per 16x16 tile, 2 waves per block.
// Fragment layouts per CDNA5 ISA 7.12.2 (wave32).
// ---------------------------------------------------------------------------
__global__ void wmma_gemm(const _Float16* __restrict__ A, int lda, long long aBatch,
                          const _Float16* __restrict__ Bt, int ldb, long long bBatch,
                          int K,
                          const float* __restrict__ bias,
                          const float* __restrict__ bng, const float* __restrict__ bnb,
                          int relu,
                          float* __restrict__ C32, int ldc,
                          _Float16* __restrict__ C16, int ldc16,
                          long long cBatch) {
  const int lane = threadIdx.x & 31;
  const int tn = blockIdx.x * 2 + (threadIdx.x >> 5);
  const int tm = blockIdx.y;
  const int bz = blockIdx.z;
  const int lh = lane >> 4;   // half-wave select
  const int lm = lane & 15;

  const _Float16* Ab = A  + (long long)bz * aBatch;
  const _Float16* Bb = Bt + (long long)bz * bBatch;
  // A lane row = tm*16+lm; K segments [8*lh,+8) and [16+8*lh,+8)
  const _Float16* ap = Ab + (long long)(tm * 16 + lm) * lda + 8 * lh;
  // B lane n-row = tn*16+lm; K segments [8*lh,+8) and [16+8*lh,+8)
  const _Float16* bp = Bb + (long long)(tn * 16 + lm) * ldb + 8 * lh;

  v8f acc = {};
  for (int k0 = 0; k0 < K; k0 += 32) {
    V16U a, b;
    a.h[0] = *(const v8h*)(ap + k0);
    a.h[1] = *(const v8h*)(ap + k0 + 16);
    b.h[0] = *(const v8h*)(bp + k0);
    b.h[1] = *(const v8h*)(bp + k0 + 16);
    acc = __builtin_amdgcn_wmma_f32_16x16x32_f16(false, a.v, false, b.v,
                                                 (short)0, acc, false, false);
  }

  const long long coff = (long long)bz * cBatch;
  const int cn = tn * 16 + lm;
  const float bv = bias ? bias[coff + cn] : 0.0f;
  float sc = 1.0f, sh = 0.0f;
  if (bng) { sc = bng[cn] * 0.9999950000374997f; sh = bnb[cn]; } // g/sqrt(1+1e-5)
#pragma unroll
  for (int r = 0; r < 8; ++r) {
    const int m = tm * 16 + 8 * lh + r;   // C row per ISA C/D layout
    float v = acc[r] + bv;
    v = v * sc + sh;
    if (relu) v = fmaxf(v, 0.0f);
    if (C32) C32[(long long)m * ldc   + coff + cn] = v;
    if (C16) C16[(long long)m * ldc16 + coff + cn] = (_Float16)v;
  }
}

// ---------------------------------------------------------------------------
// Pooling: scores[b,s] = x[b,s,:]·pool_w + pool_b   (one wave per row)
// ---------------------------------------------------------------------------
__global__ void pool_scores(const float* __restrict__ xf, const float* __restrict__ pw,
                            const float* __restrict__ pb, float* __restrict__ scores) {
  const int wave = threadIdx.x >> 5, lane = threadIdx.x & 31;
  const long long row = (long long)blockIdx.x * 8 + wave;
  const float* xr = xf + row * D_;
  float acc = 0.f;
  for (int d = lane; d < D_; d += 32) acc += xr[d] * pw[d];
  acc = warp_sum(acc);
  if (lane == 0) scores[row] = acc + pb[0];
}

// softmax over S then z[b,:] = sum_s w[b,s] x[b,s,:]   (one block of 768 / b)
__global__ void __launch_bounds__(768)
pool_reduce(const float* __restrict__ xf, const float* __restrict__ scores,
            float* __restrict__ z, _Float16* __restrict__ zh) {
  __shared__ float sw[S_];
  __shared__ float ssum;
  const int b = blockIdx.x, tid = threadIdx.x;
  if (tid < S_) sw[tid] = scores[(long long)b * S_ + tid];
  __syncthreads();
  if (tid < 32) {
    float mx = -3.4e38f;
    for (int s = tid; s < S_; s += 32) mx = fmaxf(mx, sw[s]);
    mx = warp_max(mx);
    float sum = 0.f;
    for (int s = tid; s < S_; s += 32) { float e = __expf(sw[s] - mx); sw[s] = e; sum += e; }
    sum = warp_sum(sum);
    if (tid == 0) ssum = sum;
  }
  __syncthreads();
  float acc = 0.f;
  const float* xb = xf + (long long)b * S_ * D_;
  for (int s = 0; s < S_; ++s) {
    if ((s & 7) == 0 && s + 8 < S_)
      __builtin_prefetch(xb + (long long)(s + 8) * D_ + tid, 0, 1);
    acc += sw[s] * xb[(long long)s * D_ + tid];
  }
  const float zv = acc / ssum;
  z[(long long)b * D_ + tid]  = zv;
  zh[(long long)b * D_ + tid] = (_Float16)zv;
}

// ---------------------------------------------------------------------------
// Fused attention (single query token): scores -> softmax -> x aggregation.
// One block (768 threads = 24 waves) per batch row; qk & att live in LDS.
// ---------------------------------------------------------------------------
__global__ void __launch_bounds__(768)
attn_kernel(const float* __restrict__ xf, const float* __restrict__ qk,
            _Float16* __restrict__ xa) {
  __shared__ float s_qk[H_][D_];   // 24 KB
  __shared__ float s_att[H_][S_];  // 16 KB
  const int b = blockIdx.x, tid = threadIdx.x;
  const int wave = tid >> 5, lane = tid & 31;

  for (int i = tid; i < H_ * D_; i += 768)
    s_qk[i / D_][i % D_] = qk[(long long)b * H_ * D_ + i];
  __syncthreads();

  // Phase 1: score[h,s] = (1/sqrt(96)) x[b,s,:]·qk[b,h,:]
  const float scl = 0.10206207261596575f;
  for (int s = wave; s < S_; s += 24) {
    const float* xr = xf + ((long long)b * S_ + s) * D_;
    float acc[H_] = {0, 0, 0, 0, 0, 0, 0, 0};
    for (int d = lane; d < D_; d += 32) {
      const float xv = xr[d];
#pragma unroll
      for (int h = 0; h < H_; ++h) acc[h] += xv * s_qk[h][d];
    }
#pragma unroll
    for (int h = 0; h < H_; ++h) acc[h] = warp_sum(acc[h]);
    if (lane == 0) {
#pragma unroll
      for (int h = 0; h < H_; ++h) s_att[h][s] = acc[h] * scl;
    }
  }
  __syncthreads();

  // Phase 2: softmax over s, one wave per head
  if (wave < H_) {
    const int h = wave;
    float mx = -3.4e38f;
    for (int s = lane; s < S_; s += 32) mx = fmaxf(mx, s_att[h][s]);
    mx = warp_max(mx);
    float sum = 0.f;
    for (int s = lane; s < S_; s += 32) { float e = __expf(s_att[h][s] - mx); s_att[h][s] = e; sum += e; }
    sum = warp_sum(sum);
    const float inv = 1.0f / sum;
    for (int s = lane; s < S_; s += 32) s_att[h][s] *= inv;
  }
  __syncthreads();

  // Phase 3: xa[b,h,d] = sum_s att[h,s] x[b,s,d]   (thread d = tid)
  {
    float acc[H_] = {0, 0, 0, 0, 0, 0, 0, 0};
    const float* xb = xf + (long long)b * S_ * D_;
    for (int s = 0; s < S_; ++s) {
      if ((s & 7) == 0 && s + 8 < S_)
        __builtin_prefetch(xb + (long long)(s + 8) * D_ + tid, 0, 1);
      const float xv = xb[(long long)s * D_ + tid];
#pragma unroll
      for (int h = 0; h < H_; ++h) acc[h] += s_att[h][s] * xv;
    }
#pragma unroll
    for (int h = 0; h < H_; ++h)
      xa[((long long)b * H_ + h) * D_ + tid] = (_Float16)acc[h];
  }
}

// ---------------------------------------------------------------------------
// LayerNorm 1: out = LN(z + sa[t]*z_attn)*g + b      (block 256 per row)
// ---------------------------------------------------------------------------
__global__ void ln1_kernel(const float* __restrict__ zin, const float* __restrict__ za,
                           const float* __restrict__ sa, int t,
                           const float* __restrict__ g, const float* __restrict__ bb,
                           float* __restrict__ out) {
  __shared__ float sred[8];
  const int row = blockIdx.x, tid = threadIdx.x;
  const float c = sa[t];
  float v[3]; float sum = 0.f, sq = 0.f;
#pragma unroll
  for (int i = 0; i < 3; ++i) {
    const int d = tid + i * 256;
    const float x = zin[(long long)row * D_ + d] + c * za[(long long)row * D_ + d];
    v[i] = x; sum += x; sq += x * x;
  }
  sum = block_sum256(sum, sred);
  sq  = block_sum256(sq, sred);
  const float mu = sum * (1.0f / 768.0f);
  const float var = sq * (1.0f / 768.0f) - mu * mu;
  const float rs = rsqrtf(var + 1e-5f);
#pragma unroll
  for (int i = 0; i < 3; ++i) {
    const int d = tid + i * 256;
    out[(long long)row * D_ + d] = (v[i] - mu) * rs * g[d] + bb[d];
  }
}

// LN2 + gate mix + eval-mode batchnorm, writes new z (f32 + f16)
__global__ void ln2_kernel(const float* __restrict__ z1, const float* __restrict__ zd,
                           const float* __restrict__ sd, const float* __restrict__ gate, int t,
                           const float* __restrict__ g, const float* __restrict__ bb,
                           const float* __restrict__ bng, const float* __restrict__ bnb,
                           float* __restrict__ z, _Float16* __restrict__ zh) {
  __shared__ float sred[8];
  const int row = blockIdx.x, tid = threadIdx.x;
  const float c = sd[t];
  const float gv = 1.0f / (1.0f + __expf(-gate[t]));
  float v[3]; float sum = 0.f, sq = 0.f;
#pragma unroll
  for (int i = 0; i < 3; ++i) {
    const int d = tid + i * 256;
    const float x = z1[(long long)row * D_ + d] + c * zd[(long long)row * D_ + d];
    v[i] = x; sum += x; sq += x * x;
  }
  sum = block_sum256(sum, sred);
  sq  = block_sum256(sq, sred);
  const float mu = sum * (1.0f / 768.0f);
  const float var = sq * (1.0f / 768.0f) - mu * mu;
  const float rs = rsqrtf(var + 1e-5f);
#pragma unroll
  for (int i = 0; i < 3; ++i) {
    const int d = tid + i * 256;
    const float z2 = (v[i] - mu) * rs * g[d] + bb[d];
    const float z3 = gv * z2 + (1.0f - gv) * z[(long long)row * D_ + d];
    const float zn = z3 * (bng[d] * 0.9999950000374997f) + bnb[d];
    z[(long long)row * D_ + d]  = zn;
    zh[(long long)row * D_ + d] = (_Float16)zn;
  }
}

// ---------------------------------------------------------------------------
// Spectral filter: 16-mode DFT projection, scalar complex filter, 16-mode
// real reconstruction. One block (256) per row. (m*d mod 768 keeps args small.)
// ---------------------------------------------------------------------------
__global__ void fft_kernel(const float* __restrict__ z1,
                           const float* __restrict__ fre, const float* __restrict__ fim,
                           _Float16* __restrict__ zsp) {
  __shared__ float sred[8][32];
  __shared__ float smode[32];
  const int b = blockIdx.x, tid = threadIdx.x;
  const int wave = tid >> 5, lane = tid & 31;
  float re[MODES_], im[MODES_];
#pragma unroll
  for (int m = 0; m < MODES_; ++m) { re[m] = 0.f; im[m] = 0.f; }
  const float w0 = -6.283185307179586f / 768.0f;
#pragma unroll
  for (int i = 0; i < 3; ++i) {
    const int d = tid + i * 256;
    const float x = z1[(long long)b * D_ + d];
#pragma unroll
    for (int m = 0; m < MODES_; ++m) {
      float sn, cs;
      const int md = (m * d) % 768;
      __sincosf(w0 * (float)md, &sn, &cs);
      re[m] += x * cs; im[m] += x * sn;
    }
  }
#pragma unroll
  for (int m = 0; m < MODES_; ++m) { re[m] = warp_sum(re[m]); im[m] = warp_sum(im[m]); }
  if (lane == 0) {
#pragma unroll
    for (int m = 0; m < MODES_; ++m) { sred[wave][m] = re[m]; sred[wave][16 + m] = im[m]; }
  }
  __syncthreads();
  if (tid < 32) {
    float s = 0.f;
#pragma unroll
    for (int w = 0; w < 8; ++w) s += sred[w][tid];
    smode[tid] = s;
  }
  __syncthreads();
  if (tid < 16) {          // apply complex filter F = filt * z_freq
    const float r = smode[tid], ii = smode[16 + tid];
    const float fr = fre[tid], fi = fim[tid];
    smode[tid]      = fr * r - fi * ii;
    smode[16 + tid] = fr * ii + fi * r;
  }
  __syncthreads();
  const float w1 = 6.283185307179586f / 768.0f;
#pragma unroll
  for (int i = 0; i < 3; ++i) {
    const int n = tid + i * 256;
    float acc = 0.f;
#pragma unroll
    for (int m = 0; m < MODES_; ++m) {
      float sn, cs;
      const int mn = (m * n) % 768;
      __sincosf(w1 * (float)mn, &sn, &cs);
      acc += smode[m] * cs - smode[16 + m] * sn;
    }
    zsp[(long long)b * D_ + n] = (_Float16)(acc * (1.0f / 768.0f));
  }
}

// Final classifier head (N=14 too small for WMMA) + logits accumulation (/T)
__global__ void cls_kernel(const float* __restrict__ h2, const float* __restrict__ cw,
                           const float* __restrict__ cb, float* __restrict__ out, int first) {
  const int b = blockIdx.x, lane = threadIdx.x;
  if (lane >= NCLS_) return;
  float acc = cb[lane];
  for (int k = 0; k < 128; ++k) acc += h2[(long long)b * 128 + k] * cw[k * NCLS_ + lane];
  const float v = 0.25f * acc;
  if (first) out[(long long)b * NCLS_ + lane]  = v;
  else       out[(long long)b * NCLS_ + lane] += v;
}

// Weight conversions to f16 (direct and transposed)
__global__ void cvt_f16(const float* __restrict__ src, _Float16* __restrict__ dst, long long n) {
  const long long i = (long long)blockIdx.x * blockDim.x + threadIdx.x;
  if (i < n) dst[i] = (_Float16)src[i];
}
// src (batch,R,C) -> dst (batch,C,R)
__global__ void cvt_f16_tr(const float* __restrict__ src, _Float16* __restrict__ dst,
                           int R, int C, long long n) {
  const long long i = (long long)blockIdx.x * blockDim.x + threadIdx.x;
  if (i >= n) return;
  const long long per = (long long)R * C;
  const long long bt = i / per, rem = i % per;
  const int c = (int)(rem / R), r = (int)(rem % R);
  dst[i] = (_Float16)src[bt * per + (long long)r * C + c];
}

// filt[t,m] = mean_d(scale_real) + i*mean_d(scale_imag)
__global__ void filt_kernel(const float* __restrict__ sr, const float* __restrict__ si,
                            float* __restrict__ fre, float* __restrict__ fim) {
  const int tm = blockIdx.x, lane = threadIdx.x;
  const float* a  = sr + (long long)tm * D_;
  const float* bb = si + (long long)tm * D_;
  float ar = 0.f, ai = 0.f;
  for (int d = lane; d < D_; d += 32) { ar += a[d]; ai += bb[d]; }
  ar = warp_sum(ar); ai = warp_sum(ai);
  if (lane == 0) { fre[tm] = ar * (1.0f / 768.0f); fim[tm] = ai * (1.0f / 768.0f); }
}

// ---------------------------------------------------------------------------
extern "C" void kernel_launch(void* const* d_in, const int* in_sizes, int n_in,
                              void* d_out, int out_size, void* d_ws, size_t ws_size,
                              hipStream_t stream) {
  (void)in_sizes; (void)n_in; (void)out_size; (void)ws_size;
  const float* x_feat    = (const float*)d_in[0];
  const float* pool_w    = (const float*)d_in[1];
  const float* pool_b    = (const float*)d_in[2];
  const float* w_q       = (const float*)d_in[3];
  const float* w_k       = (const float*)d_in[4];
  const float* w_v       = (const float*)d_in[5];
  const float* b_qkv     = (const float*)d_in[6];
  const float* w_o       = (const float*)d_in[7];
  const float* b_o       = (const float*)d_in[8];
  const float* ln1_g     = (const float*)d_in[9];
  const float* ln1_b     = (const float*)d_in[10];
  const float* ln2_g     = (const float*)d_in[11];
  const float* ln2_b     = (const float*)d_in[12];
  const float* scale_real= (const float*)d_in[13];
  const float* scale_imag= (const float*)d_in[14];
  const float* proj_w    = (const float*)d_in[15];
  const float* proj_b    = (const float*)d_in[16];
  const float* bn_g      = (const float*)d_in[17];
  const float* bn_b      = (const float*)d_in[18];
  const float* gate      = (const float*)d_in[19];
  const float* scale_attn    = (const float*)d_in[20];
  const float* scale_denoise = (const float*)d_in[21];
  const float* f1_w      = (const float*)d_in[22];
  const float* f1_b      = (const float*)d_in[23];
  const float* f1_bn_g   = (const float*)d_in[24];
  const float* f1_bn_b   = (const float*)d_in[25];
  const float* f2_w      = (const float*)d_in[26];
  const float* f2_b      = (const float*)d_in[27];
  const float* f2_bn_g   = (const float*)d_in[28];
  const float* f2_bn_b   = (const float*)d_in[29];
  const float* cls_w     = (const float*)d_in[30];
  const float* cls_b     = (const float*)d_in[31];
  float* out = (float*)d_out;

  // Workspace layout (256B aligned slices)
  char* base = (char*)d_ws;
  size_t off = 0;
  auto alloc = [&](size_t bytes) -> void* {
    off = (off + 255) & ~(size_t)255;
    void* p = base + off;
    off += bytes;
    return p;
  };
  const long long DD = (long long)D_ * D_;
  _Float16* wq_h   = (_Float16*)alloc((size_t)T_ * DD * 2);
  _Float16* wkT_h  = (_Float16*)alloc((size_t)T_ * DD * 2);
  _Float16* wv_h   = (_Float16*)alloc((size_t)T_ * DD * 2);
  _Float16* wo_h   = (_Float16*)alloc((size_t)T_ * DD * 2);
  _Float16* proj_h = (_Float16*)alloc((size_t)T_ * DD * 2);
  _Float16* f1T_h  = (_Float16*)alloc((size_t)768 * 256 * 2);
  _Float16* f2T_h  = (_Float16*)alloc((size_t)256 * 128 * 2);
  float*    fre    = (float*)alloc((size_t)T_ * MODES_ * 4);
  float*    fim    = (float*)alloc((size_t)T_ * MODES_ * 4);
  float*    scores = (float*)alloc((size_t)B_ * S_ * 4);
  float*    z      = (float*)alloc((size_t)B_ * D_ * 4);
  _Float16* zh     = (_Float16*)alloc((size_t)B_ * D_ * 2);
  _Float16* q_h    = (_Float16*)alloc((size_t)B_ * D_ * 2);
  float*    qk     = (float*)alloc((size_t)B_ * H_ * D_ * 4);
  _Float16* xa_h   = (_Float16*)alloc((size_t)B_ * H_ * D_ * 2);
  _Float16* o_h    = (_Float16*)alloc((size_t)B_ * D_ * 2);
  float*    z_attn = (float*)alloc((size_t)B_ * D_ * 4);
  float*    z1     = (float*)alloc((size_t)B_ * D_ * 4);
  _Float16* zsp_h  = (_Float16*)alloc((size_t)B_ * D_ * 2);
  float*    z_den  = (float*)alloc((size_t)B_ * D_ * 4);
  _Float16* h1_h   = (_Float16*)alloc((size_t)B_ * 256 * 2);
  float*    h2     = (float*)alloc((size_t)B_ * 128 * 4);

  // ---- weight prep (f16 + required transposes) ----
  {
    const long long n = (long long)T_ * DD;
    const unsigned g = (unsigned)((n + 255) / 256);
    cvt_f16<<<g, 256, 0, stream>>>(w_q,    wq_h,   n);
    cvt_f16<<<g, 256, 0, stream>>>(w_v,    wv_h,   n);
    cvt_f16<<<g, 256, 0, stream>>>(w_o,    wo_h,   n);
    cvt_f16<<<g, 256, 0, stream>>>(proj_w, proj_h, n);
    cvt_f16_tr<<<g, 256, 0, stream>>>(w_k, wkT_h, D_, D_, n);
    const long long n1 = 768LL * 256;
    cvt_f16_tr<<<(unsigned)((n1 + 255) / 256), 256, 0, stream>>>(f1_w, f1T_h, 768, 256, n1);
    const long long n2 = 256LL * 128;
    cvt_f16_tr<<<(unsigned)((n2 + 255) / 256), 256, 0, stream>>>(f2_w, f2T_h, 256, 128, n2);
  }
  filt_kernel<<<T_ * MODES_, 32, 0, stream>>>(scale_real, scale_imag, fre, fim);

  // ---- attention pooling ----
  pool_scores<<<B_ * S_ / 8, 256, 0, stream>>>(x_feat, pool_w, pool_b, scores);
  pool_reduce<<<B_, 768, 0, stream>>>(x_feat, scores, z, zh);

  // ---- trunk ----
  for (int t = 0; t < T_; ++t) {
    const _Float16* wq_t   = wq_h   + (long long)t * DD;
    const _Float16* wkT_t  = wkT_h  + (long long)t * DD;
    const _Float16* wv_t   = wv_h   + (long long)t * DD;
    const _Float16* wo_t   = wo_h   + (long long)t * DD;
    const _Float16* proj_t = proj_h + (long long)t * DD;

    // q = z @ w_q^T + bq                         (256x768x768)
    wmma_gemm<<<dim3(24, 16, 1), 64, 0, stream>>>(
        zh, D_, 0LL, wq_t, D_, 0LL, D_,
        b_qkv + (long long)t * 3 * D_, nullptr, nullptr, 0,
        nullptr, 0, q_h, D_, 0LL);

    // qk[b,h,:] = q_h @ w_k[h-block]             (256x768x96, batched over h)
    wmma_gemm<<<dim3(24, 16, H_), 64, 0, stream>>>(
        q_h, D_, (long long)DH_, wkT_t, D_, (long long)DH_, DH_,
        nullptr, nullptr, nullptr, 0,
        qk, H_ * D_, nullptr, 0, (long long)D_);

    // fused score -> softmax -> x aggregation (2 streaming passes over x_feat)
    attn_kernel<<<B_, 768, 0, stream>>>(x_feat, qk, xa_h);

    // o[b, h*96:+96] = xa[b,h,:] @ w_v[h-block]^T + bv   (256x96x768, batched)
    wmma_gemm<<<dim3(3, 16, H_), 64, 0, stream>>>(
        xa_h, H_ * D_, (long long)D_, wv_t, D_, (long long)DH_ * D_, D_,
        b_qkv + (long long)t * 3 * D_ + 2 * D_, nullptr, nullptr, 0,
        nullptr, 0, o_h, D_, (long long)DH_);

    // z_attn = o @ w_o^T + b_o
    wmma_gemm<<<dim3(24, 16, 1), 64, 0, stream>>>(
        o_h, D_, 0LL, wo_t, D_, 0LL, D_,
        b_o + (long long)t * D_, nullptr, nullptr, 0,
        z_attn, D_, nullptr, 0, 0LL);

    ln1_kernel<<<B_, 256, 0, stream>>>(z, z_attn, scale_attn, t,
                                       ln1_g + (long long)t * D_, ln1_b + (long long)t * D_, z1);

    fft_kernel<<<B_, 256, 0, stream>>>(z1, fre + t * MODES_, fim + t * MODES_, zsp_h);

    // z_den = z_spatial @ proj_w^T + proj_b
    wmma_gemm<<<dim3(24, 16, 1), 64, 0, stream>>>(
        zsp_h, D_, 0LL, proj_t, D_, 0LL, D_,
        proj_b + (long long)t * D_, nullptr, nullptr, 0,
        z_den, D_, nullptr, 0, 0LL);

    ln2_kernel<<<B_, 256, 0, stream>>>(z1, z_den, scale_denoise, gate, t,
                                       ln2_g + (long long)t * D_, ln2_b + (long long)t * D_,
                                       bn_g + (long long)t * D_, bn_b + (long long)t * D_,
                                       z, zh);

    // classifier: h1 = relu(bn(z@f1+b)), h2 = relu(bn(h1@f2+b))
    wmma_gemm<<<dim3(8, 16, 1), 64, 0, stream>>>(
        zh, D_, 0LL, f1T_h, D_, 0LL, D_,
        f1_b, f1_bn_g, f1_bn_b, 1,
        nullptr, 0, h1_h, 256, 0LL);
    wmma_gemm<<<dim3(4, 16, 1), 64, 0, stream>>>(
        h1_h, 256, 0LL, f2T_h, 256, 0LL, 256,
        f2_b, f2_bn_g, f2_bn_b, 1,
        h2, 128, nullptr, 0, 0LL);

    cls_kernel<<<B_, 32, 0, stream>>>(h2, cls_w, cls_b, out, t == 0 ? 1 : 0);
  }
}
